// ALoraLinear_498216206382
// MI455X (gfx1250) — compile-verified, matrix operand order checked by reading, not run on hardware
//
#include <hip/hip_runtime.h>

typedef float  v2f  __attribute__((ext_vector_type(2)));
typedef float  v4f  __attribute__((ext_vector_type(4)));
typedef float  v8f  __attribute__((ext_vector_type(8)));
typedef __bf16 v4bf __attribute__((ext_vector_type(4)));
typedef __bf16 v8bf __attribute__((ext_vector_type(8)));
typedef __bf16 v16bf __attribute__((ext_vector_type(16)));

// fp32 WMMA: D = A(16x4) * B(4x16) + C
#define WMMA_F32(a, b, c) \
  __builtin_amdgcn_wmma_f32_16x16x4_f32(false, (a), false, (b), (short)0, (c), false, false)
// bf16 WMMA: D = A(16x32) * B(32x16) + C (fp32 accumulate)
#define WMMA_BF16(a, b, c) \
  __builtin_amdgcn_wmma_f32_16x16x32_bf16(false, (a), false, (b), (short)0, (c), false, false)

constexpr int S_LEN  = 4096;
constexpr int D_IN   = 4096;
constexpr int D_OUT  = 4096;
constexpr int RANK   = 32;
constexpr int BATCH  = 4;
constexpr int M_TOT  = BATCH * S_LEN;          // 16384 rows
constexpr float SCALING = 2.0f;
constexpr int LDSP  = 20;   // fp32 LDS row stride (floats): 80B
constexpr int LDSB  = 40;   // bf16 LDS row stride (bf16):   80B, 16B-aligned, conflict-free
constexpr int TILE_BYTES = 128 * LDSB * 2;     // 10240B per bf16 tile
constexpr unsigned AHI_OFF = 0 * TILE_BYTES;
constexpr unsigned ALO_OFF = 1 * TILE_BYTES;
constexpr unsigned BHI_OFF = 2 * TILE_BYTES;
constexpr unsigned BLO_OFF = 3 * TILE_BYTES;
constexpr unsigned BUF_BYTES = 4 * TILE_BYTES; // 40960B per double-buffer half

// --- CDNA5 async global->LDS copy (ASYNCcnt path), GVS addressing ----------
__device__ __forceinline__ void async_b128_to_lds(unsigned lds_off,
                                                  const void* sbase,
                                                  unsigned voff) {
    asm volatile("global_load_async_to_lds_b128 %0, %1, %2"
                 :: "v"(lds_off), "v"(voff),
                    "s"((unsigned long long)(__UINTPTR_TYPE__)sbase)
                 : "memory");
}
__device__ __forceinline__ void wait_async0() {
#if __has_builtin(__builtin_amdgcn_s_wait_asynccnt)
    __builtin_amdgcn_s_wait_asynccnt(0);
#else
    asm volatile("s_wait_asynccnt 0x0" ::: "memory");
#endif
}

// ---------------------------------------------------------------------------
// Kernel 0: elementwise fp32 -> (bf16 hi, bf16 lo) split, 4 elems/thread.
// ---------------------------------------------------------------------------
__global__ __launch_bounds__(256) void split_bf16_kernel(
    const float* __restrict__ src, __bf16* __restrict__ hi,
    __bf16* __restrict__ lo, int n)
{
    int i = (blockIdx.x * 256 + threadIdx.x) * 4;
    if (i >= n) return;
    v4f v = *(const v4f*)(src + i);
    v4bf h, l;
    #pragma unroll
    for (int j = 0; j < 4; ++j) {
        h[j] = (__bf16)v[j];
        l[j] = (__bf16)(v[j] - (float)h[j]);
    }
    *(v4bf*)(hi + i) = h;
    *(v4bf*)(lo + i) = l;
}

// ---------------------------------------------------------------------------
// Kernel 1: T[m, r] = mask(m) * SCALING * sum_d x[m,d] * A[r,d]
// Memory-bound -> exact fp32 WMMA. Also emits bf16 hi/lo copies of T (when
// workspace allows) so the fully-presplit main kernel can async-load the
// LoRA tail chunk too.
// ---------------------------------------------------------------------------
__global__ __launch_bounds__(256) void alora_xa_kernel(
    const float* __restrict__ X, const int* __restrict__ offsets,
    const float* __restrict__ A, float* __restrict__ T,
    __bf16* __restrict__ Thi, __bf16* __restrict__ Tlo)
{
    __shared__ float Xs[128][LDSP];
    __shared__ float Aw[32][LDSP];

    const int tid  = threadIdx.x;
    const int wave = tid >> 5;
    const int lane = tid & 31;
    const int lh   = lane >> 4;
    const int l15  = lane & 15;
    const int mBase = blockIdx.x * 128;

    const int b    = mBase >> 12;               // block fits in one sample
    int koff = offsets[b];
    if (koff > S_LEN) koff = S_LEN;
    const int tail = S_LEN - koff;

    v8f acc0 = 0.0f, acc1 = 0.0f;

    for (int kc = 0; kc < D_IN; kc += 16) {
        __syncthreads();
        #pragma unroll
        for (int i = 0; i < 2; ++i) {
            int id  = tid + i * 256;
            int row = id >> 2, c4 = (id & 3) * 4;
            *(v4f*)&Xs[row][c4] =
                *(const v4f*)(X + (size_t)(mBase + row) * D_IN + kc + c4);
        }
        if (tid < 128) {
            int row = tid >> 2, c4 = (tid & 3) * 4;
            *(v4f*)&Aw[row][c4] =
                *(const v4f*)(A + (size_t)row * D_IN + kc + c4);
        }
        __syncthreads();

        #pragma unroll
        for (int kk = 0; kk < 16; kk += 4) {
            v2f a  = *(const v2f*)&Xs[wave * 16 + l15][kk + lh * 2];
            v2f b0 = *(const v2f*)&Aw[l15][kk + lh * 2];
            v2f b1 = *(const v2f*)&Aw[16 + l15][kk + lh * 2];
            acc0 = WMMA_F32(a, b0, acc0);
            acc1 = WMMA_F32(a, b1, acc1);
        }
    }

    #pragma unroll
    for (int i = 0; i < 8; ++i) {
        int m = mBase + wave * 16 + i + lh * 8;
        int s = m & (S_LEN - 1);
        float sc = (s >= tail) ? SCALING : 0.0f;
        float v0 = acc0[i] * sc, v1 = acc1[i] * sc;
        T[(size_t)m * RANK + l15]      = v0;
        T[(size_t)m * RANK + 16 + l15] = v1;
        if (Thi) {                              // uniform branch
            __bf16 h0 = (__bf16)v0, h1 = (__bf16)v1;
            Thi[(size_t)m * RANK + l15]      = h0;
            Thi[(size_t)m * RANK + 16 + l15] = h1;
            Tlo[(size_t)m * RANK + l15]      = (__bf16)(v0 - (float)h0);
            Tlo[(size_t)m * RANK + 16 + l15] = (__bf16)(v1 - (float)h1);
        }
    }
}

// ---------------------------------------------------------------------------
// Kernel 2: out = [x | T] @ [W | B_w]^T + bias, split-precision bf16x3
// (hi*hi + hi*lo + lo*hi, fp32 accumulate; ~2^-16 product error).
// Double-buffered LDS (one barrier/chunk); PB/PX select async global->LDS
// B128 DMA for pre-split bf16 operands vs on-the-fly fp32->hi/lo VALU split.
// PB&&PX => pure WMMA + async-DMA hot loop, zero conversion VALU.
// 128x128 tile, BK=32, 8 waves (2x4), each wave 4x2 tiles x3 WMMAs.
// ---------------------------------------------------------------------------
template<bool PB, bool PX>
__global__ __launch_bounds__(256) void alora_main_kernel(
    const float* __restrict__ X, const float* __restrict__ W,
    const float* __restrict__ bias, const float* __restrict__ Bw,
    const float* __restrict__ T,
    const __bf16* __restrict__ Whi, const __bf16* __restrict__ Wlo,
    const __bf16* __restrict__ Bwhi, const __bf16* __restrict__ Bwlo,
    const __bf16* __restrict__ Xhi, const __bf16* __restrict__ Xlo,
    const __bf16* __restrict__ Thi, const __bf16* __restrict__ Tlo,
    float* __restrict__ out)
{
    // one LDS symbol -> pool starts at LDS offset 0; raw offsets usable in asm
    __shared__ alignas(16) unsigned char ldsPool[2 * BUF_BYTES];

    const int tid  = threadIdx.x;
    const int wave = tid >> 5;
    const int lane = tid & 31;
    const int lh   = lane >> 4;
    const int l15  = lane & 15;
    const int wm   = wave >> 2;     // 0..1 -> row offset wm*64
    const int wn   = wave & 3;      // 0..3 -> col offset wn*32
    const int mBase = blockIdx.y * 128;
    const int nBase = blockIdx.x * 128;

    v8f acc[4][2];
    #pragma unroll
    for (int tm = 0; tm < 4; ++tm)
        #pragma unroll
        for (int tn = 0; tn < 2; ++tn)
            acc[tm][tn] = 0.0f;

    // ---- stage one BK=32 chunk into LDS buffer `base` ----
    auto loadTiles = [&](int kc, unsigned base) {
        const bool mainK = kc < D_IN;
        // B side (W / B_w)
        if constexpr (PB) {
            const __bf16* hbase = mainK ? Whi : Bwhi;
            const __bf16* lbase = mainK ? Wlo : Bwlo;
            const unsigned rowB = mainK ? (unsigned)(D_IN * 2) : (unsigned)(RANK * 2);
            const unsigned colB = mainK ? (unsigned)(kc * 2) : 0u;
            #pragma unroll
            for (int i = 0; i < 2; ++i) {
                int id  = tid + i * 256;        // 0..511
                int row = id >> 2, c16 = id & 3;
                unsigned voff = (unsigned)(nBase + row) * rowB + colB + c16 * 16;
                unsigned loff = base + row * (LDSB * 2) + c16 * 16;
                async_b128_to_lds(BHI_OFF + loff, hbase, voff);
                async_b128_to_lds(BLO_OFF + loff, lbase, voff);
            }
        } else {
            auto BHI = (__bf16(*)[LDSB])(ldsPool + base + BHI_OFF);
            auto BLO = (__bf16(*)[LDSB])(ldsPool + base + BLO_OFF);
            const float* srcB = mainK ? (W  + (size_t)nBase * D_IN + kc)
                                      : (Bw + (size_t)nBase * RANK);
            const size_t str  = mainK ? (size_t)D_IN : (size_t)RANK;
            #pragma unroll
            for (int i = 0; i < 4; ++i) {
                int id  = tid + i * 256;
                int row = id >> 3, c4 = (id & 7) * 4;
                v4f vb = *(const v4f*)(srcB + (size_t)row * str + c4);
                v4bf bh, bl;
                #pragma unroll
                for (int j = 0; j < 4; ++j) {
                    bh[j] = (__bf16)vb[j];
                    bl[j] = (__bf16)(vb[j] - (float)bh[j]);
                }
                *(v4bf*)&BHI[row][c4] = bh;
                *(v4bf*)&BLO[row][c4] = bl;
            }
        }
        // A side (x / T)
        if constexpr (PX) {
            const __bf16* hbase = mainK ? Xhi : Thi;
            const __bf16* lbase = mainK ? Xlo : Tlo;
            const unsigned rowA = mainK ? (unsigned)(D_IN * 2) : (unsigned)(RANK * 2);
            const unsigned colA = mainK ? (unsigned)(kc * 2) : 0u;
            #pragma unroll
            for (int i = 0; i < 2; ++i) {
                int id  = tid + i * 256;
                int row = id >> 2, c16 = id & 3;
                unsigned voff = (unsigned)(mBase + row) * rowA + colA + c16 * 16;
                unsigned loff = base + row * (LDSB * 2) + c16 * 16;
                async_b128_to_lds(AHI_OFF + loff, hbase, voff);
                async_b128_to_lds(ALO_OFF + loff, lbase, voff);
            }
        } else {
            auto AHI = (__bf16(*)[LDSB])(ldsPool + base + AHI_OFF);
            auto ALO = (__bf16(*)[LDSB])(ldsPool + base + ALO_OFF);
            const float* srcA = mainK ? (X + (size_t)mBase * D_IN + kc)
                                      : (T + (size_t)mBase * RANK);
            const size_t str  = mainK ? (size_t)D_IN : (size_t)RANK;
            #pragma unroll
            for (int i = 0; i < 4; ++i) {
                int id  = tid + i * 256;        // 0..1023
                int row = id >> 3, c4 = (id & 7) * 4;
                v4f va = *(const v4f*)(srcA + (size_t)row * str + c4);
                v4bf ah, al;
                #pragma unroll
                for (int j = 0; j < 4; ++j) {
                    ah[j] = (__bf16)va[j];
                    al[j] = (__bf16)(va[j] - (float)ah[j]);
                }
                *(v4bf*)&AHI[row][c4] = ah;
                *(v4bf*)&ALO[row][c4] = al;
            }
        }
    };

    // ---- fragments (ISA 7.12.2 16-bit A 16x32 layout) + 24 WMMAs ----
    auto computeTiles = [&](unsigned base) {
        auto AHI = (__bf16(*)[LDSB])(ldsPool + base + AHI_OFF);
        auto ALO = (__bf16(*)[LDSB])(ldsPool + base + ALO_OFF);
        auto BHI = (__bf16(*)[LDSB])(ldsPool + base + BHI_OFF);
        auto BLO = (__bf16(*)[LDSB])(ldsPool + base + BLO_OFF);
        v16bf fah[4], fal[4], fbh[2], fbl[2];
        #pragma unroll
        for (int tm = 0; tm < 4; ++tm) {
            int row = wm * 64 + tm * 16 + l15;
            *(v8bf*)&fah[tm]       = *(const v8bf*)&AHI[row][lh * 8];
            *((v8bf*)&fah[tm] + 1) = *(const v8bf*)&AHI[row][16 + lh * 8];
            *(v8bf*)&fal[tm]       = *(const v8bf*)&ALO[row][lh * 8];
            *((v8bf*)&fal[tm] + 1) = *(const v8bf*)&ALO[row][16 + lh * 8];
        }
        #pragma unroll
        for (int tn = 0; tn < 2; ++tn) {
            int n = wn * 32 + tn * 16 + l15;
            *(v8bf*)&fbh[tn]       = *(const v8bf*)&BHI[n][lh * 8];
            *((v8bf*)&fbh[tn] + 1) = *(const v8bf*)&BHI[n][16 + lh * 8];
            *(v8bf*)&fbl[tn]       = *(const v8bf*)&BLO[n][lh * 8];
            *((v8bf*)&fbl[tn] + 1) = *(const v8bf*)&BLO[n][16 + lh * 8];
        }
        #pragma unroll
        for (int tm = 0; tm < 4; ++tm)
            #pragma unroll
            for (int tn = 0; tn < 2; ++tn) {
                acc[tm][tn] = WMMA_BF16(fal[tm], fbh[tn], acc[tm][tn]);
                acc[tm][tn] = WMMA_BF16(fah[tm], fbl[tn], acc[tm][tn]);
                acc[tm][tn] = WMMA_BF16(fah[tm], fbh[tn], acc[tm][tn]);
            }
    };

    // ---- double-buffered main loop: one barrier per chunk ----
    constexpr int NCHUNK = (D_IN + RANK) / 32;  // 129
    unsigned curB = 0, nxtB = BUF_BYTES;
    loadTiles(0, curB);
    for (int chunk = 0; chunk < NCHUNK; ++chunk) {
        if constexpr (PB || PX) wait_async0();  // my async LDS writes done
        __syncthreads();                        // everyone's writes/reads fenced
        if (chunk + 1 < NCHUNK) loadTiles((chunk + 1) * 32, nxtB);
        computeTiles(curB);
        unsigned t = curB; curB = nxtB; nxtB = t;
    }

    float bb[2];
    #pragma unroll
    for (int tn = 0; tn < 2; ++tn)
        bb[tn] = bias[nBase + wn * 32 + tn * 16 + l15];

    #pragma unroll
    for (int tm = 0; tm < 4; ++tm)
        #pragma unroll
        for (int tn = 0; tn < 2; ++tn)
            #pragma unroll
            for (int i = 0; i < 8; ++i) {
                int row = mBase + wm * 64 + tm * 16 + i + lh * 8;
                int col = nBase + wn * 32 + tn * 16 + l15;
                out[(size_t)row * D_OUT + col] = acc[tm][tn][i] + bb[tn];
            }
}

// ---------------------------------------------------------------------------
extern "C" void kernel_launch(void* const* d_in, const int* in_sizes, int n_in,
                              void* d_out, int out_size, void* d_ws, size_t ws_size,
                              hipStream_t stream) {
    (void)in_sizes; (void)n_in; (void)out_size;
    const float* x   = (const float*)d_in[0];   // [4, 4096, 4096]
    const int*   off = (const int*)d_in[1];     // [4]
    const float* W   = (const float*)d_in[2];   // [4096, 4096]
    const float* b   = (const float*)d_in[3];   // [4096]
    const float* A   = (const float*)d_in[4];   // [32, 4096]
    const float* Bw  = (const float*)d_in[5];   // [4096, 32]
    float* out = (float*)d_out;                 // [4, 4096, 4096]

    // workspace layout: T | Thi | Tlo | Whi | Wlo | Bwhi | Bwlo | Xhi | Xlo
    const size_t T_ELEMS  = (size_t)M_TOT * RANK;           // 512K
    const size_t W_ELEMS  = (size_t)D_OUT * D_IN;           // 16.7M
    const size_t BW_ELEMS = (size_t)D_OUT * RANK;           // 131K
    const size_t X_ELEMS  = (size_t)M_TOT * D_IN;           // 67.1M
    const size_t NEED_W    = T_ELEMS * 4 + 2 * T_ELEMS * 2
                           + 2 * W_ELEMS * 2 + 2 * BW_ELEMS * 2;   // ~72 MiB
    const size_t NEED_FULL = NEED_W + 2 * X_ELEMS * 2;             // ~340 MiB

    char* ws = (char*)d_ws;
    float*  T    = (float*)ws;
    __bf16* Thi  = (__bf16*)(ws + T_ELEMS * 4);
    __bf16* Tlo  = Thi + T_ELEMS;
    __bf16* Whi  = Tlo + T_ELEMS;
    __bf16* Wlo  = Whi + W_ELEMS;
    __bf16* Bwhi = Wlo + W_ELEMS;
    __bf16* Bwlo = Bwhi + BW_ELEMS;
    __bf16* Xhi  = Bwlo + BW_ELEMS;
    __bf16* Xlo  = Xhi + X_ELEMS;

    const bool tierW = ws_size >= NEED_W;
    const bool tierF = ws_size >= NEED_FULL;

    alora_xa_kernel<<<M_TOT / 128, 256, 0, stream>>>(
        x, off, A, T, tierW ? Thi : nullptr, tierW ? Tlo : nullptr);

    dim3 grid(D_OUT / 128, M_TOT / 128);        // 32 x 128 blocks

    if (tierF) {
        split_bf16_kernel<<<(int)(W_ELEMS / 1024), 256, 0, stream>>>(W, Whi, Wlo, (int)W_ELEMS);
        split_bf16_kernel<<<(int)(BW_ELEMS / 1024), 256, 0, stream>>>(Bw, Bwhi, Bwlo, (int)BW_ELEMS);
        split_bf16_kernel<<<(int)(X_ELEMS / 1024), 256, 0, stream>>>(x, Xhi, Xlo, (int)X_ELEMS);
        alora_main_kernel<true, true><<<grid, 256, 0, stream>>>(
            x, W, b, Bw, T, Whi, Wlo, Bwhi, Bwlo, Xhi, Xlo, Thi, Tlo, out);
    } else if (tierW) {
        split_bf16_kernel<<<(int)(W_ELEMS / 1024), 256, 0, stream>>>(W, Whi, Wlo, (int)W_ELEMS);
        split_bf16_kernel<<<(int)(BW_ELEMS / 1024), 256, 0, stream>>>(Bw, Bwhi, Bwlo, (int)BW_ELEMS);
        alora_main_kernel<true, false><<<grid, 256, 0, stream>>>(
            x, W, b, Bw, T, Whi, Wlo, Bwhi, Bwlo, nullptr, nullptr, nullptr, nullptr, out);
    } else {
        alora_main_kernel<false, false><<<grid, 256, 0, stream>>>(
            x, W, b, Bw, T, nullptr, nullptr, nullptr, nullptr,
            nullptr, nullptr, nullptr, nullptr, out);
    }
}
